// ScaledDotProduct_76501957476888
// MI455X (gfx1250) — compile-verified, matrix operand order checked by reading, not run on hardware
//
#include <hip/hip_runtime.h>

#define S_LEN 2048
#define D_DIM 128
#define NBH   32           // B*H
#define BN    32           // key block
#define WAVES 8
#define BM    (WAVES * 16) // 128 query rows per workgroup
#define NT    (S_LEN / BN)

typedef __attribute__((ext_vector_type(16))) __bf16 v16bf;
typedef __attribute__((ext_vector_type(8)))  float  v8f;

union Frag { v16bf bf; unsigned int u[8]; };

__device__ __forceinline__ unsigned int pack2bf_sw(float lo, float hi) {
  unsigned int a = __float_as_uint(lo), b = __float_as_uint(hi);
  a = (a + 0x7FFFu + ((a >> 16) & 1u)) >> 16;
  b = (b + 0x7FFFu + ((b >> 16) & 1u)) & 0xFFFF0000u;
  return (a & 0xFFFFu) | b;
}
__device__ __forceinline__ unsigned int cvt2bf(float lo, float hi) {
#if __has_builtin(__builtin_amdgcn_cvt_pk_bf16_f32)
  typedef __attribute__((ext_vector_type(2))) __bf16 v2bf;
  union { v2bf v; unsigned int u; } cv;
  cv.v = __builtin_amdgcn_cvt_pk_bf16_f32(lo, hi);
  return cv.u;
#else
  return pack2bf_sw(lo, hi);
#endif
}
__device__ __forceinline__ float fexp2(float x) {
#if __has_builtin(__builtin_amdgcn_exp2f)
  return __builtin_amdgcn_exp2f(x);
#else
  return exp2f(x);
#endif
}
// per-lane K-index pattern for 16-bit 16x32 A/B fragments (ISA 7.12.2)
__device__ __forceinline__ int kpat(int vv, int half) {
  return ((vv & 4) << 2) + (half << 3) + ((vv & 3) << 1);
}
__device__ __forceinline__ float rmax16(float x) {
  #pragma unroll
  for (int m = 8; m >= 1; m >>= 1) x = fmaxf(x, __shfl_xor(x, m, 32));
  return x;
}
__device__ __forceinline__ void load_kfrag(Frag& f, const unsigned int* kb,
                                           int tile, int c, int ln16, int half) {
  #pragma unroll
  for (int vv = 0; vv < 8; ++vv)
    f.u[vv] = kb[(tile * 16 + ln16) * (D_DIM / 2) + ((c * 32 + kpat(vv, half)) >> 1)];
}
__device__ __forceinline__ void load_vfrag(Frag& f, const unsigned int* vtp,
                                           int dc, int ln16, int half) {
  #pragma unroll
  for (int vv = 0; vv < 8; ++vv)
    f.u[vv] = vtp[(dc * 16 + ln16) * (BN / 2) + (kpat(vv, half) >> 1)];
}

__global__ __launch_bounds__(256)
void ScaledDotProduct_fa_bf16_wmma(const float* __restrict__ q,
                                   const float* __restrict__ k,
                                   const float* __restrict__ v,
                                   float* __restrict__ out) {
  __shared__ unsigned int   ldsK[2][BN][D_DIM / 2];  // bf16 pairs K[key][d]  (16 KB)
  __shared__ unsigned short ldsVT[2][D_DIM][BN];     // V^T [d][pslot]        (16 KB)
  __shared__ unsigned int   ldsP[WAVES][16][16];     // P [row][keypair]      ( 8 KB)

  const int t    = threadIdx.x;
  const int lane = t & 31;
  const int wv   = t >> 5;
  const int half = lane >> 4;
  const int ln16 = lane & 15;
  const size_t base = (size_t)blockIdx.y * S_LEN * D_DIM;
  const int q0 = blockIdx.x * BM + wv * 16;

  const float* kp = k + base;
  const float* vp = v + base;

  // staging coords: 8 threads per key row, 16 cols each
  const int srow  = t >> 3;
  const int seg   = (t & 7) * 16;
  const int pslot = ((srow & 15) << 1) | (srow >> 4);  // interleaved key slot

  // ---- Q fragments with softmax scale * log2(e) folded in ----
  const float SCL = 0.12751744f;  // (1/sqrt(128)) * log2(e)
  Frag qa[4];
  {
    const float* qrow = q + base + (size_t)(q0 + ln16) * D_DIM;
    #pragma unroll
    for (int c = 0; c < 4; ++c)
      #pragma unroll
      for (int vv = 0; vv < 8; ++vv) {
        int d = c * 32 + kpat(vv, half);
        qa[c].u[vv] = cvt2bf(qrow[d] * SCL, qrow[d + 1] * SCL);
      }
  }

  // all-ones B fragment: rowsum(P) = P x ones via one WMMA
  Frag ones;
  #pragma unroll
  for (int vv = 0; vv < 8; ++vv) ones.u[vv] = 0x3F803F80u;  // bf16 1.0 pair

  float m_i[8], l_i[8];
  v8f o[8];
  #pragma unroll
  for (int r = 0; r < 8; ++r) { m_i[r] = -3.0e38f; l_i[r] = 0.f; }
  #pragma unroll
  for (int dc = 0; dc < 8; ++dc) o[dc] = (v8f){};

  // ---- prologue: stage tile 0 into buffer 0 ----
  {
    const float4* kr = (const float4*)(kp + (size_t)srow * D_DIM + seg);
    const float4* vr = (const float4*)(vp + (size_t)srow * D_DIM + seg);
    float4 kq[4], vq[4];
    #pragma unroll
    for (int j = 0; j < 4; ++j) { kq[j] = kr[j]; vq[j] = vr[j]; }
    const float* kf = (const float*)kq;
    const float* vf = (const float*)vq;
    #pragma unroll
    for (int j = 0; j < 8; ++j)
      ldsK[0][srow][(seg >> 1) + j] = cvt2bf(kf[2 * j], kf[2 * j + 1]);
    #pragma unroll
    for (int j = 0; j < 8; ++j) {
      unsigned int u = cvt2bf(vf[2 * j], vf[2 * j + 1]);
      ldsVT[0][seg + 2 * j][pslot]     = (unsigned short)u;
      ldsVT[0][seg + 2 * j + 1][pslot] = (unsigned short)(u >> 16);
    }
  }

  for (int it = 0; it < NT; ++it) {
    const int cur = it & 1, nxt = cur ^ 1;
    __syncthreads();   // buf[cur] staged; prior readers of buf[nxt] done

    // ---- issue next tile's global loads (latency hidden behind WMMAs) ----
    float4 kq[4], vq[4];
    const bool more = (it + 1 < NT);
    if (more) {
      const float4* kr =
          (const float4*)(kp + (size_t)(it + 1) * BN * D_DIM + (size_t)srow * D_DIM + seg);
      const float4* vr =
          (const float4*)(vp + (size_t)(it + 1) * BN * D_DIM + (size_t)srow * D_DIM + seg);
      #pragma unroll
      for (int j = 0; j < 4; ++j) { kq[j] = kr[j]; vq[j] = vr[j]; }
    }

    // ---- S = Q K^T : two 16x16 tiles, ping-pong K fragments ----
    const unsigned int* kbase = &ldsK[cur][0][0];
    v8f s0 = (v8f){}, s1 = (v8f){};
    {
      Frag kf0, kf1;
      load_kfrag(kf0, kbase, 0, 0, ln16, half);
      #pragma unroll
      for (int st = 0; st < 8; ++st) {
        const int c = st >> 1;
        if (st + 1 < 8) {
          if ((st + 1) & 1) load_kfrag(kf1, kbase, 1, (st + 1) >> 1, ln16, half);
          else              load_kfrag(kf0, kbase, 0, (st + 1) >> 1, ln16, half);
        }
        if (st & 1)
          s1 = __builtin_amdgcn_wmma_f32_16x16x32_bf16(false, qa[c].bf, false, kf1.bf,
                                                       (short)0, s1, false, false);
        else
          s0 = __builtin_amdgcn_wmma_f32_16x16x32_bf16(false, qa[c].bf, false, kf0.bf,
                                                       (short)0, s0, false, false);
      }
    }

    // ---- online softmax part 1: row max (half-wave reduce) + exp2 ----
    float alpha[8];
    #pragma unroll
    for (int r = 0; r < 8; ++r) {
      float mx = rmax16(fmaxf(s0[r], s1[r]));
      float mn = fmaxf(m_i[r], mx);
      alpha[r] = fexp2(m_i[r] - mn);
      s0[r] = fexp2(s0[r] - mn);
      s1[r] = fexp2(s1[r] - mn);
      m_i[r] = mn;
    }

    // ---- convert + store next K/V tile now (frees staging registers) ----
    if (more) {
      const float* kf = (const float*)kq;
      const float* vf = (const float*)vq;
      #pragma unroll
      for (int j = 0; j < 8; ++j)
        ldsK[nxt][srow][(seg >> 1) + j] = cvt2bf(kf[2 * j], kf[2 * j + 1]);
      #pragma unroll
      for (int j = 0; j < 8; ++j) {
        unsigned int u = cvt2bf(vf[2 * j], vf[2 * j + 1]);
        ldsVT[nxt][seg + 2 * j][pslot]     = (unsigned short)u;
        ldsVT[nxt][seg + 2 * j + 1][pslot] = (unsigned short)(u >> 16);
      }
    }

    // ---- P: C-layout -> packed bf16 A-layout (interleaved key slots) ----
    #pragma unroll
    for (int r = 0; r < 8; ++r)
      ldsP[wv][r + (half << 3)][ln16] = cvt2bf(s0[r], s1[r]);
    asm volatile("s_wait_dscnt 0" ::: "memory");   // same-wave LDS store->load
    Frag pa;
    {
      const unsigned int* pw = &ldsP[wv][0][0];
      #pragma unroll
      for (int vv = 0; vv < 8; ++vv)
        pa.u[vv] = pw[ln16 * 16 + (kpat(vv, half) >> 1)];
    }

    const unsigned int* vtp = (const unsigned int*)&ldsVT[cur][0][0];
    Frag vf0, vf1;
    load_vfrag(vf0, vtp, 0, ln16, half);   // in flight during lsum + rescale

    // ---- row sums via ones-WMMA: lsum[r] = sum_k P[row][k] ----
    v8f lsum = __builtin_amdgcn_wmma_f32_16x16x32_bf16(false, pa.bf, false, ones.bf,
                                                       (short)0, (v8f){}, false, false);
    #pragma unroll
    for (int r = 0; r < 8; ++r) l_i[r] = l_i[r] * alpha[r] + lsum[r];
    #pragma unroll
    for (int dc = 0; dc < 8; ++dc)
      #pragma unroll
      for (int r = 0; r < 8; ++r) o[dc][r] *= alpha[r];

    // ---- O += P V : 8 WMMAs, ping-pong V fragments ----
    #pragma unroll
    for (int dc = 0; dc < 8; ++dc) {
      if (dc + 1 < 8) {
        if ((dc + 1) & 1) load_vfrag(vf1, vtp, dc + 1, ln16, half);
        else              load_vfrag(vf0, vtp, dc + 1, ln16, half);
      }
      if (dc & 1)
        o[dc] = __builtin_amdgcn_wmma_f32_16x16x32_bf16(false, pa.bf, false, vf1.bf,
                                                        (short)0, o[dc], false, false);
      else
        o[dc] = __builtin_amdgcn_wmma_f32_16x16x32_bf16(false, pa.bf, false, vf0.bf,
                                                        (short)0, o[dc], false, false);
    }
  }

  // ---- epilogue: normalize and store fp32 ----
  #pragma unroll
  for (int r = 0; r < 8; ++r) {
    float inv = 1.0f / l_i[r];
    float* orow = out + base + (size_t)(q0 + r + (half << 3)) * D_DIM;
    #pragma unroll
    for (int dc = 0; dc < 8; ++dc) orow[dc * 16 + ln16] = o[dc][r] * inv;
  }
}

extern "C" void kernel_launch(void* const* d_in, const int* in_sizes, int n_in,
                              void* d_out, int out_size, void* d_ws, size_t ws_size,
                              hipStream_t stream) {
  const float* q = (const float*)d_in[0];
  const float* k = (const float*)d_in[1];
  const float* v = (const float*)d_in[2];
  float* out = (float*)d_out;
  dim3 grid(S_LEN / BM, NBH, 1);
  dim3 block(WAVES * 32, 1, 1);
  ScaledDotProduct_fa_bf16_wmma<<<grid, block, 0, stream>>>(q, k, v, out);
}
